// E2ETTTBlock_12764642804272
// MI455X (gfx1250) — matrix operand order repaired and checked by reading, not compiled, not run on hardware
//
#include <hip/hip_runtime.h>
#include <hip/hip_bf16.h>

// ---------------- problem constants ----------------
constexpr int B  = 2;
constexpr int N  = 2048;
constexpr int H  = 1024;
constexpr int NH = 16;
constexpr int HD = 64;
constexpr int I  = 4096;
constexpr float EPS = 1e-6f;
constexpr int MTOT = B * N;          // 4096 token rows

typedef _Float16 h16;
typedef __attribute__((ext_vector_type(16))) _Float16 v16h;
typedef __attribute__((ext_vector_type(8)))  _Float16 v8h;
typedef __attribute__((ext_vector_type(8)))  float    v8f;

union V16 { v16h v; v8h h[2]; };

__device__ __forceinline__ v16h load_v16(const h16* p) {
    V16 u;
    u.h[0] = *(const v8h*)(p);
    u.h[1] = *(const v8h*)(p + 8);
    return u.v;
}

// A-fragment load: lane row = m-row, halves j -> K = hi*16 + j
__device__ __forceinline__ v16h load_afrag(const h16* row, int k0, int hi) {
    V16 u;
    u.h[0] = *(const v8h*)(row + k0 + hi * 16);
    u.h[1] = *(const v8h*)(row + k0 + hi * 16 + 8);
    return u.v;
}

// ---------------- block reduction helper ----------------
__device__ __forceinline__ float block_reduce_sum(float v) {
    __shared__ float sm[8];
    #pragma unroll
    for (int m = 16; m >= 1; m >>= 1) v += __shfl_xor(v, m, 32);
    const int wave = threadIdx.x >> 5;
    const int lane = threadIdx.x & 31;
    const int nw = blockDim.x >> 5;
    if (lane == 0) sm[wave] = v;
    __syncthreads();
    float tot = 0.f;
    for (int i = 0; i < nw; ++i) tot += sm[i];
    __syncthreads();   // guard against reuse of sm by a later call
    return tot;
}

// ---------------- elementwise kernels ----------------
__global__ void cvt_f32_f16(const float* __restrict__ in, h16* __restrict__ out, int n) {
    for (int i = blockIdx.x * blockDim.x + threadIdx.x; i < n; i += gridDim.x * blockDim.x)
        out[i] = (h16)in[i];
}

// out_h = rmsnorm(in, w) as f16; one block per row
__global__ void rmsnorm_to_f16(const float* __restrict__ in, const float* __restrict__ w,
                               h16* __restrict__ out, int C) {
    const size_t row = blockIdx.x;
    const float* x = in + row * C;
    float ss = 0.f;
    for (int i = threadIdx.x; i < C; i += blockDim.x) { float v = x[i]; ss += v * v; }
    ss = block_reduce_sum(ss);
    const float r = rsqrtf(ss / (float)C + EPS);
    for (int i = threadIdx.x; i < C; i += blockDim.x)
        out[row * C + i] = (h16)(x[i] * r * w[i]);
}

// x1 = x + rmsnorm(attn_out, w_post); m_in = f16(rmsnorm(x1, w_mlp))
__global__ void post_attn(const float* __restrict__ attn_out, const float* __restrict__ x,
                          const float* __restrict__ w_post, const float* __restrict__ w_mlp,
                          float* __restrict__ x1, h16* __restrict__ m_in, int C) {
    const size_t row = blockIdx.x;
    float ss = 0.f;
    for (int i = threadIdx.x; i < C; i += blockDim.x) {
        float v = attn_out[row * C + i]; ss += v * v;
    }
    ss = block_reduce_sum(ss);
    const float r1 = rsqrtf(ss / (float)C + EPS);
    float ss2 = 0.f;
    for (int i = threadIdx.x; i < C; i += blockDim.x) {
        float t = x[row * C + i] + attn_out[row * C + i] * r1 * w_post[i];
        x1[row * C + i] = t;
        ss2 += t * t;
    }
    ss2 = block_reduce_sum(ss2);
    const float r2 = rsqrtf(ss2 / (float)C + EPS);
    for (int i = threadIdx.x; i < C; i += blockDim.x)
        m_in[row * C + i] = (h16)(x1[row * C + i] * r2 * w_mlp[i]);
}

// out = x1 + rmsnorm(m, w)
__global__ void final_residual(const float* __restrict__ m, const float* __restrict__ x1,
                               const float* __restrict__ w, float* __restrict__ out, int C) {
    const size_t row = blockIdx.x;
    float ss = 0.f;
    for (int i = threadIdx.x; i < C; i += blockDim.x) { float v = m[row * C + i]; ss += v * v; }
    ss = block_reduce_sum(ss);
    const float r = rsqrtf(ss / (float)C + EPS);
    for (int i = threadIdx.x; i < C; i += blockDim.x)
        out[row * C + i] = x1[row * C + i] + m[row * C + i] * r * w[i];
}

// g = f16( silu(h1) * h3 )
__global__ void silu_mul(const float* __restrict__ h1, const float* __restrict__ h3,
                         h16* __restrict__ g, int n) {
    for (int i = blockIdx.x * blockDim.x + threadIdx.x; i < n; i += gridDim.x * blockDim.x) {
        float a = h1[i];
        float s = a / (1.f + __expf(-a));
        g[i] = (h16)(s * h3[i]);
    }
}

// ---------------- WMMA GEMM: C(M,Nc) f32 = A(M,K) f16 @ W(Nc,K)^T f16 ----------------
// block = 256 threads (8 waves). Block tile 256(M) x 64(N); per wave 32 x 64 (2x4 WMMA
// tiles). K-step 32, double-buffered fragment sets so loads overlap WMMA issue.
__global__ void __launch_bounds__(256) gemm_f16_nt(float* __restrict__ C,
                                                   const h16* __restrict__ A,
                                                   const h16* __restrict__ W,
                                                   int M, int Nc, int K) {
    const int wave = threadIdx.x >> 5;
    const int lane = threadIdx.x & 31;
    const int lane16 = lane & 15;
    const int hi = lane >> 4;
    const int m0 = blockIdx.x * 256 + wave * 32;
    const int n0 = blockIdx.y * 64;

    const h16* arow0 = A + (size_t)(m0 + lane16) * K;
    const h16* arow1 = A + (size_t)(m0 + 16 + lane16) * K;
    const h16* wb    = W + ((size_t)n0 + lane16) * K;

    v8f acc[2][4] = {};

    // two explicit fragment sets (ping-pong), all in named registers
    v16h a0_s0, a1_s0, b0_s0, b1_s0, b2_s0, b3_s0;
    v16h a0_s1, a1_s1, b0_s1, b1_s1, b2_s1, b3_s1;

    auto load_set0 = [&](int k0) {
        a0_s0 = load_afrag(arow0, k0, hi);
        a1_s0 = load_afrag(arow1, k0, hi);
        b0_s0 = load_v16(wb + (size_t)0 * 16 * K + k0 + hi * 16);
        b1_s0 = load_v16(wb + (size_t)1 * 16 * K + k0 + hi * 16);
        b2_s0 = load_v16(wb + (size_t)2 * 16 * K + k0 + hi * 16);
        b3_s0 = load_v16(wb + (size_t)3 * 16 * K + k0 + hi * 16);
    };
    auto load_set1 = [&](int k0) {
        a0_s1 = load_afrag(arow0, k0, hi);
        a1_s1 = load_afrag(arow1, k0, hi);
        b0_s1 = load_v16(wb + (size_t)0 * 16 * K + k0 + hi * 16);
        b1_s1 = load_v16(wb + (size_t)1 * 16 * K + k0 + hi * 16);
        b2_s1 = load_v16(wb + (size_t)2 * 16 * K + k0 + hi * 16);
        b3_s1 = load_v16(wb + (size_t)3 * 16 * K + k0 + hi * 16);
    };
    auto compute0 = [&]() {
        acc[0][0] = __builtin_amdgcn_wmma_f32_16x16x32_f16(false, a0_s0, false, b0_s0, (short)0, acc[0][0], false, false);
        acc[0][1] = __builtin_amdgcn_wmma_f32_16x16x32_f16(false, a0_s0, false, b1_s0, (short)0, acc[0][1], false, false);
        acc[0][2] = __builtin_amdgcn_wmma_f32_16x16x32_f16(false, a0_s0, false, b2_s0, (short)0, acc[0][2], false, false);
        acc[0][3] = __builtin_amdgcn_wmma_f32_16x16x32_f16(false, a0_s0, false, b3_s0, (short)0, acc[0][3], false, false);
        acc[1][0] = __builtin_amdgcn_wmma_f32_16x16x32_f16(false, a1_s0, false, b0_s0, (short)0, acc[1][0], false, false);
        acc[1][1] = __builtin_amdgcn_wmma_f32_16x16x32_f16(false, a1_s0, false, b1_s0, (short)0, acc[1][1], false, false);
        acc[1][2] = __builtin_amdgcn_wmma_f32_16x16x32_f16(false, a1_s0, false, b2_s0, (short)0, acc[1][2], false, false);
        acc[1][3] = __builtin_amdgcn_wmma_f32_16x16x32_f16(false, a1_s0, false, b3_s0, (short)0, acc[1][3], false, false);
    };
    auto compute1 = [&]() {
        acc[0][0] = __builtin_amdgcn_wmma_f32_16x16x32_f16(false, a0_s1, false, b0_s1, (short)0, acc[0][0], false, false);
        acc[0][1] = __builtin_amdgcn_wmma_f32_16x16x32_f16(false, a0_s1, false, b1_s1, (short)0, acc[0][1], false, false);
        acc[0][2] = __builtin_amdgcn_wmma_f32_16x16x32_f16(false, a0_s1, false, b2_s1, (short)0, acc[0][2], false, false);
        acc[0][3] = __builtin_amdgcn_wmma_f32_16x16x32_f16(false, a0_s1, false, b3_s1, (short)0, acc[0][3], false, false);
        acc[1][0] = __builtin_amdgcn_wmma_f32_16x16x32_f16(false, a1_s1, false, b0_s1, (short)0, acc[1][0], false, false);
        acc[1][1] = __builtin_amdgcn_wmma_f32_16x16x32_f16(false, a1_s1, false, b1_s1, (short)0, acc[1][1], false, false);
        acc[1][2] = __builtin_amdgcn_wmma_f32_16x16x32_f16(false, a1_s1, false, b2_s1, (short)0, acc[1][2], false, false);
        acc[1][3] = __builtin_amdgcn_wmma_f32_16x16x32_f16(false, a1_s1, false, b3_s1, (short)0, acc[1][3], false, false);
    };

    load_set0(0);
    // K is a multiple of 64 for all GEMMs in this block (1024 or 4096)
    for (int k0 = 0; k0 < K; k0 += 64) {
        load_set1(k0 + 32);      // prefetch while computing set0
        compute0();
        if (k0 + 64 < K) load_set0(k0 + 64);   // prefetch while computing set1
        compute1();
    }

    // D layout: VGPR r holds row m = hi*8 + r, col n = lane16
    #pragma unroll
    for (int mt = 0; mt < 2; ++mt) {
        float* crow = C + (size_t)(m0 + mt * 16 + hi * 8) * Nc + n0 + lane16;
        #pragma unroll
        for (int r = 0; r < 8; ++r) {
            crow[(size_t)r * Nc +  0] = acc[mt][0][r];
            crow[(size_t)r * Nc + 16] = acc[mt][1][r];
            crow[(size_t)r * Nc + 32] = acc[mt][2][r];
            crow[(size_t)r * Nc + 48] = acc[mt][3][r];
        }
    }
}

// ---------------- QKV post: per-head rmsnorm + RoPE, write q/k (b,h,n,d), v^T (b,h,d,n) ----------------
__global__ void __launch_bounds__(64) qkv_post(const float* __restrict__ qkv,
                                               const float* __restrict__ qw,
                                               const float* __restrict__ kw,
                                               h16* __restrict__ qh, h16* __restrict__ kh,
                                               h16* __restrict__ vt) {
    const int d = threadIdx.x;            // 0..63
    const int wave = d >> 5;
    const int lane = d & 31;
    const int idx = blockIdx.x;           // (b*N + n)*NH + head
    const int head = idx & (NH - 1);
    const int bn = idx >> 4;              // b*N + n
    const int b = bn >> 11;               // N = 2048
    const int n = bn & (N - 1);

    const float* base = qkv + (size_t)bn * (3 * H) + head * HD;
    const float qv = base[d];
    const float kv = base[H + d];
    const float vv = base[2 * H + d];

    __shared__ float red[2];
    __shared__ float sq[HD], sk[HD];

    float s = qv * qv;
    #pragma unroll
    for (int m = 16; m >= 1; m >>= 1) s += __shfl_xor(s, m, 32);
    if (lane == 0) red[wave] = s;
    __syncthreads();
    const float ssq = red[0] + red[1];
    __syncthreads();
    s = kv * kv;
    #pragma unroll
    for (int m = 16; m >= 1; m >>= 1) s += __shfl_xor(s, m, 32);
    if (lane == 0) red[wave] = s;
    __syncthreads();
    const float ssk = red[0] + red[1];

    const float qn = qv * rsqrtf(ssq / (float)HD + EPS) * qw[d];
    const float kn = kv * rsqrtf(ssk / (float)HD + EPS) * kw[d];
    sq[d] = qn; sk[d] = kn;
    __syncthreads();

    const int fi = d & 31;                                  // freq index
    const float LOG_THETA = 9.210340371976184f;             // ln(10000)
    const float inv_freq = __expf(-((float)(2 * fi) / (float)HD) * LOG_THETA);
    const float ang = (float)n * inv_freq;
    const float c = cosf(ang), sn = sinf(ang);
    const float qrot = (d < 32) ? -sq[d + 32] : sq[d - 32];
    const float krot = (d < 32) ? -sk[d + 32] : sk[d - 32];
    const float qo = qn * c + qrot * sn;
    const float ko = kn * c + krot * sn;

    const size_t bh = (size_t)b * NH + head;
    const size_t qoff = (bh * N + n) * HD + d;
    qh[qoff] = (h16)qo;
    kh[qoff] = (h16)ko;
    vt[(bh * HD + d) * N + n] = (h16)vv;
}

// ---------------- flash attention (causal; WIN>=N so window is no-op) ----------------
// block = 256 threads (8 waves); each wave: 16 q rows; key tiles of 32.
__global__ void __launch_bounds__(256) attn_kernel(h16* __restrict__ o_h,
                                                   const h16* __restrict__ q_h,
                                                   const h16* __restrict__ k_h,
                                                   const h16* __restrict__ vt_h) {
    __shared__ h16 lds_p[8][16 * 32];
    const int wave = threadIdx.x >> 5;
    const int lane = threadIdx.x & 31;
    const int lane16 = lane & 15;
    const int hi = lane >> 4;
    const int qb = blockIdx.x & 15;        // N/128 = 16 q-blocks
    const int bh = blockIdx.x >> 4;        // b*NH + h
    const int b = bh >> 4;
    const int h = bh & (NH - 1);
    const int q0 = qb * 128 + wave * 16;

    const h16* qbase = q_h + ((size_t)bh * N + q0) * HD;
    const h16* kbase = k_h + (size_t)bh * N * HD;
    const h16* vbase = vt_h + (size_t)bh * HD * N;

    // Q fragments for both head-dim halves (K-steps), reused for all key tiles
    v16h aq[2];
    {
        const h16* qrow = qbase + (size_t)lane16 * HD;
        #pragma unroll
        for (int kk = 0; kk < 2; ++kk)
            aq[kk] = load_afrag(qrow, kk * 32, hi);
    }

    float mi[8], li[8];
    v8f o[4] = {};
    #pragma unroll
    for (int r = 0; r < 8; ++r) { mi[r] = -1e30f; li[r] = 0.f; }

    const int kend = q0 + 16;
    for (int kb = 0; kb < kend; kb += 32) {
        // batch all K-fragment loads, then issue the 4 S WMMAs
        v16h bk00, bk01, bk10, bk11;
        {
            const h16* krow0 = kbase + (size_t)(kb + lane16) * HD;
            const h16* krow1 = kbase + (size_t)(kb + 16 + lane16) * HD;
            bk00 = load_v16(krow0 + hi * 16);
            bk01 = load_v16(krow0 + 32 + hi * 16);
            bk10 = load_v16(krow1 + hi * 16);
            bk11 = load_v16(krow1 + 32 + hi * 16);
        }
        v8f sacc[2] = {};
        sacc[0] = __builtin_amdgcn_wmma_f32_16x16x32_f16(false, aq[0], false, bk00, (short)0, sacc[0], false, false);
        sacc[0] = __builtin_amdgcn_wmma_f32_16x16x32_f16(false, aq[1], false, bk01, (short)0, sacc[0], false, false);
        sacc[1] = __builtin_amdgcn_wmma_f32_16x16x32_f16(false, aq[0], false, bk10, (short)0, sacc[1], false, false);
        sacc[1] = __builtin_amdgcn_wmma_f32_16x16x32_f16(false, aq[1], false, bk11, (short)0, sacc[1], false, false);

        // prefetch V fragments for this key tile (consumed after softmax)
        v16h bv0, bv1, bv2, bv3;
        {
            const h16* vrow = vbase + (size_t)lane16 * N + kb + hi * 16;
            bv0 = load_v16(vrow + (size_t)0  * 16 * N);
            bv1 = load_v16(vrow + (size_t)1  * 16 * N);
            bv2 = load_v16(vrow + (size_t)2  * 16 * N);
            bv3 = load_v16(vrow + (size_t)3  * 16 * N);
        }

        // mask + online softmax (rows live in 16-lane half-waves)
        float sc0[8], sc1[8], p0[8], p1[8];
        #pragma unroll
        for (int r = 0; r < 8; ++r) {
            const int qi = q0 + hi * 8 + r;
            float a0 = sacc[0][r] * 0.125f;
            float a1 = sacc[1][r] * 0.125f;
            if (kb + lane16 > qi)      a0 = -1e30f;
            if (kb + 16 + lane16 > qi) a1 = -1e30f;
            sc0[r] = a0; sc1[r] = a1;
        }
        #pragma unroll
        for (int r = 0; r < 8; ++r) {
            float t = fmaxf(sc0[r], sc1[r]);
            #pragma unroll
            for (int m = 8; m >= 1; m >>= 1) t = fmaxf(t, __shfl_xor(t, m, 32));
            const float mnew = fmaxf(mi[r], t);
            const float scale = __expf(mi[r] - mnew);
            mi[r] = mnew;
            p0[r] = __expf(sc0[r] - mnew);
            p1[r] = __expf(sc1[r] - mnew);
            float rs = p0[r] + p1[r];
            #pragma unroll
            for (int m = 8; m >= 1; m >>= 1) rs += __shfl_xor(rs, m, 32);
            li[r] = li[r] * scale + rs;
            #pragma unroll
            for (int t4 = 0; t4 < 4; ++t4) o[t4][r] *= scale;
        }
        // transpose P into an A-fragment via per-wave LDS tile (16 rows x 32 keys)
        h16* pl = lds_p[wave];
        #pragma unroll
        for (int r = 0; r < 8; ++r) {
            pl[(hi * 8 + r) * 32 + lane16]      = (h16)p0[r];
            pl[(hi * 8 + r) * 32 + 16 + lane16] = (h16)p1[r];
        }
        asm volatile("s_wait_dscnt 0" ::: "memory");
        v16h ap;
        {
            const h16* pr = pl + (size_t)lane16 * 32;
            V16 u;
            u.h[0] = *(const v8h*)(pr + hi * 16);
            u.h[1] = *(const v8h*)(pr + hi * 16 + 8);
            ap = u.v;
        }
        // O += P @ V  (V^T rows are contiguous over keys)
        o[0] = __builtin_amdgcn_wmma_f32_16x16x32_f16(false, ap, false, bv0, (short)0, o[0], false, false);
        o[1] = __builtin_amdgcn_wmma_f32_16x16x32_f16(false, ap, false, bv1, (short)0, o[1], false, false);
        o[2] = __builtin_amdgcn_wmma_f32_16x16x32_f16(false, ap, false, bv2, (short)0, o[2], false, false);
        o[3] = __builtin_amdgcn_wmma_f32_16x16x32_f16(false, ap, false, bv3, (short)0, o[3], false, false);
    }
    // epilogue: normalize and write o in (b, n, H) layout as f16
    h16* orow = o_h + ((size_t)b * N + q0 + hi * 8) * H + h * HD + lane16;
    #pragma unroll
    for (int r = 0; r < 8; ++r) {
        const float inv = 1.f / li[r];
        orow[(size_t)r * H +  0] = (h16)(o[0][r] * inv);
        orow[(size_t)r * H + 16] = (h16)(o[1][r] * inv);
        orow[(size_t)r * H + 32] = (h16)(o[2][r] * inv);
        orow[(size_t)r * H + 48] = (h16)(o[3][r] * inv);
    }
}

// ---------------- workspace layout (bytes) ----------------
constexpr size_t MB = 1ull << 20;
constexpr size_t OFF_WQ   = 0;          //  6 MB  qkv_w f16  (3H*H)
constexpr size_t OFF_WO   = 6  * MB;    //  2 MB  out_w f16
constexpr size_t OFF_W1   = 8  * MB;    //  8 MB  w1 f16
constexpr size_t OFF_W2   = 16 * MB;    //  8 MB  w2 f16
constexpr size_t OFF_W3   = 24 * MB;    //  8 MB  w3 f16
constexpr size_t OFF_SEQ  = 32 * MB;    //  8 MB  seq_in_h / m_in_h (f16, 4096x1024)
constexpr size_t OFF_A    = 40 * MB;    // 64 MB  qkv_f32 / h1_f32 / m_f32
constexpr size_t OFF_QH   = 104 * MB;   //  8 MB  q f16 (region reused as h3 later)
constexpr size_t OFF_KH   = 112 * MB;   //  8 MB  k f16 (region reused as h3 later)
constexpr size_t OFF_VT   = 120 * MB;   //  8 MB  v^T f16 (region reused as h3 later)
constexpr size_t OFF_OH   = 128 * MB;   //  8 MB  attn out f16 (region reused as h3 later)
constexpr size_t OFF_B    = 104 * MB;   // h3_f32 (64 MB), reuses q/k/vt/o region
constexpr size_t OFF_C    = 168 * MB;   // 32 MB  attn_out_f32 (16) / g f16 (32)
constexpr size_t OFF_X1   = 200 * MB;   // 16 MB  x1 f32

extern "C" void kernel_launch(void* const* d_in, const int* in_sizes, int n_in,
                              void* d_out, int out_size, void* d_ws, size_t ws_size,
                              hipStream_t stream) {
    const float* x        = (const float*)d_in[0];
    const float* qkv_w    = (const float*)d_in[1];
    const float* out_w    = (const float*)d_in[2];
    const float* w1       = (const float*)d_in[3];
    const float* w2       = (const float*)d_in[4];
    const float* w3       = (const float*)d_in[5];
    const float* q_norm_w = (const float*)d_in[6];
    const float* k_norm_w = (const float*)d_in[7];
    const float* attn_n_w = (const float*)d_in[8];
    const float* post_n_w = (const float*)d_in[9];
    const float* mlp_n_w  = (const float*)d_in[10];
    const float* ffn_n_w  = (const float*)d_in[11];
    float* out = (float*)d_out;

    char* ws = (char*)d_ws;
    h16*   wq_h   = (h16*)(ws + OFF_WQ);
    h16*   wo_h   = (h16*)(ws + OFF_WO);
    h16*   w1_h   = (h16*)(ws + OFF_W1);
    h16*   w2_h   = (h16*)(ws + OFF_W2);
    h16*   w3_h   = (h16*)(ws + OFF_W3);
    h16*   seq_h  = (h16*)(ws + OFF_SEQ);   // also m_in_h
    float* bufA   = (float*)(ws + OFF_A);   // qkv / h1 / m
    h16*   q_hb   = (h16*)(ws + OFF_QH);
    h16*   k_hb   = (h16*)(ws + OFF_KH);
    h16*   vt_hb  = (h16*)(ws + OFF_VT);
    h16*   o_hb   = (h16*)(ws + OFF_OH);
    float* bufB   = (float*)(ws + OFF_B);   // h3
    float* attn_f = (float*)(ws + OFF_C);
    h16*   g_h    = (h16*)(ws + OFF_C);     // reuses region C after attn_f is dead
    float* x1_f   = (float*)(ws + OFF_X1);

    const dim3 blk256(256), blk64(64);

    // weights -> f16
    cvt_f32_f16<<<4096, blk256, 0, stream>>>(qkv_w, wq_h, 3 * H * H);
    cvt_f32_f16<<<2048, blk256, 0, stream>>>(out_w, wo_h, H * H);
    cvt_f32_f16<<<4096, blk256, 0, stream>>>(w1, w1_h, I * H);
    cvt_f32_f16<<<4096, blk256, 0, stream>>>(w2, w2_h, H * I);
    cvt_f32_f16<<<4096, blk256, 0, stream>>>(w3, w3_h, I * H);

    // seq_in = rmsnorm(x, attn_norm_w)  (f16)
    rmsnorm_to_f16<<<MTOT, blk256, 0, stream>>>(x, attn_n_w, seq_h, H);

    // qkv = seq_in @ qkv_w^T   (4096 x 3072, K=1024)
    gemm_f16_nt<<<dim3(MTOT / 256, (3 * H) / 64), blk256, 0, stream>>>(
        bufA, seq_h, wq_h, MTOT, 3 * H, H);

    // qk-norm + RoPE + transpose
    qkv_post<<<MTOT * NH, blk64, 0, stream>>>(bufA, q_norm_w, k_norm_w, q_hb, k_hb, vt_hb);

    // flash attention
    attn_kernel<<<B * NH * (N / 128), blk256, 0, stream>>>(o_hb, q_hb, k_hb, vt_hb);

    // attn_out = o @ out_w^T   (4096 x 1024, K=1024)
    gemm_f16_nt<<<dim3(MTOT / 256, H / 64), blk256, 0, stream>>>(
        attn_f, o_hb, wo_h, MTOT, H, H);

    // x1 = x + rmsnorm(attn_out); m_in = f16(rmsnorm(x1))
    post_attn<<<MTOT, blk256, 0, stream>>>(attn_f, x, post_n_w, mlp_n_w, x1_f, seq_h, H);

    // h1 = m_in @ w1^T ; h3 = m_in @ w3^T   (4096 x 4096, K=1024)
    gemm_f16_nt<<<dim3(MTOT / 256, I / 64), blk256, 0, stream>>>(bufA, seq_h, w1_h, MTOT, I, H);
    gemm_f16_nt<<<dim3(MTOT / 256, I / 64), blk256, 0, stream>>>(bufB, seq_h, w3_h, MTOT, I, H);

    // g = f16(silu(h1) * h3)
    silu_mul<<<8192, blk256, 0, stream>>>(bufA, bufB, g_h, MTOT * I);

    // m = g @ w2^T   (4096 x 1024, K=4096)
    gemm_f16_nt<<<dim3(MTOT / 256, H / 64), blk256, 0, stream>>>(bufA, g_h, w2_h, MTOT, H, I);

    // out = x1 + rmsnorm(m, ffn_post_norm_w)
    final_residual<<<MTOT, blk256, 0, stream>>>(bufA, x1_f, ffn_n_w, out, H);
}
